// MHA_Layer_84524956385871
// MI455X (gfx1250) — compile-verified, matrix operand order checked by reading, not run on hardware
//
#include <hip/hip_runtime.h>

// ---------------------------------------------------------------------------
// MHA layer for MI455X (gfx1250): bf16 WMMA everywhere, fp32 accumulation.
// B=2, S=2048, H=1024, 16 heads x 64. Flash-style online softmax so no
// S x S score matrix is ever materialized. GEMM tiles are staged into LDS
// with gfx1250 async-LDS loads (ASYNCcnt) and double-buffered so the DMA
// overlaps the WMMA stream.
// ---------------------------------------------------------------------------

#define H_DIM    1024
#define NHEAD    16
#define HEAD_DIM 64
#define SEQ      2048
#define BATCH    2
#define TOKENS   (BATCH * SEQ)   // 4096

typedef __bf16 bf16;
typedef __attribute__((ext_vector_type(16))) __bf16 v16bf;
typedef __attribute__((ext_vector_type(4)))  __bf16 v4bf;
typedef __attribute__((ext_vector_type(8)))  float  v8f;

__device__ __forceinline__ v8f wmma_bf16(v16bf a, v16bf b, v8f c) {
  // D = A(16x32 bf16) * B(32x16 bf16) + C(16x16 f32)
  return __builtin_amdgcn_wmma_f32_16x16x32_bf16(
      /*neg_a=*/false, a, /*neg_b=*/false, b,
      /*c_mod=*/(short)0, c, /*reuse_a=*/false, /*reuse_b=*/false);
}

// gfx1250 async copy: 16 bytes global -> LDS per lane, tracked by ASYNCcnt.
// ldsaddr = per-lane LDS byte offset (low 32 bits of the generic pointer).
__device__ __forceinline__ void async_ld_b128(unsigned ldsaddr, const bf16* gptr) {
  asm volatile("global_load_async_to_lds_b128 %0, %1, off"
               :: "v"(ldsaddr), "v"(gptr)
               : "memory");
}
__device__ __forceinline__ void wait_async0() {
  asm volatile("s_wait_asynccnt 0x0" ::: "memory");
}

// ---------------------------------------------------------------------------
// fp32 -> bf16 conversion (vectorized x4)
// ---------------------------------------------------------------------------
__global__ __launch_bounds__(256)
void cvt_f32_bf16(const float* __restrict__ in, bf16* __restrict__ out, int n4) {
  int i = blockIdx.x * 256 + threadIdx.x;
  if (i >= n4) return;
  float4 f = ((const float4*)in)[i];
  v4bf o;
  o[0] = (bf16)f.x; o[1] = (bf16)f.y; o[2] = (bf16)f.z; o[3] = (bf16)f.w;
  ((v4bf*)out)[i] = o;
}

// ---------------------------------------------------------------------------
// GEMM: C[M][N] = A[M][K] @ W[N][K]^T + bias.  (torch Linear convention)
// Block: 256 threads = 8 waves, tile 128(M) x 128(N), K-step 32.
// Tiles staged via async-LDS loads into a double buffer; each iteration
// issues the next tile's DMA before running this tile's 8 WMMAs per wave.
// A-frag lane: row = m0 + w*16 + (lane&15), 16 contiguous K at (lane>>4)*16.
// B-frag lane: row of W (= output col) = n0 + nt*16 + (lane&15), same K half.
// Consistent K-permutation in A and B fragments => correct dot products.
// ---------------------------------------------------------------------------
template <bool OUT_F32>
__global__ __launch_bounds__(256)
void gemm_xwT(const bf16* __restrict__ A, const bf16* __restrict__ W,
              const float* __restrict__ bias, void* __restrict__ Cout,
              int Ktot, int Ntot) {
  __shared__ bf16 As[2][128][40];   // row stride 80B (16B-aligned reads)
  __shared__ bf16 Ws[2][128][40];

  const int tid  = threadIdx.x;
  const int wv   = tid >> 5;
  const int lane = tid & 31;
  const int ln   = lane & 15;
  const int lh   = lane >> 4;
  const int m0   = blockIdx.x * 128;
  const int n0   = blockIdx.y * 128;

  // cooperative fill: each thread DMAs 32B of A and 32B of W per K-step
  const int lrow  = tid >> 1;
  const int lhalf = tid & 1;
  const bf16* ag = A + (size_t)(m0 + lrow) * Ktot + lhalf * 16;
  const bf16* wg = W + (size_t)(n0 + lrow) * Ktot + lhalf * 16;

  // per-lane LDS byte offsets (generic-pointer low 32 bits = LDS offset)
  const unsigned asb = (unsigned)(size_t)(const void*)&As[0][lrow][lhalf * 16];
  const unsigned wsb = (unsigned)(size_t)(const void*)&Ws[0][lrow][lhalf * 16];
  const unsigned BUFSTRIDE = 128 * 40 * sizeof(bf16);   // 10240 B

  v8f zero = {};
  v8f acc[8];
#pragma unroll
  for (int i = 0; i < 8; ++i) acc[i] = zero;

  // prologue: DMA tile 0 into buffer 0
  async_ld_b128(asb,      ag);
  async_ld_b128(asb + 16, ag + 8);
  async_ld_b128(wsb,      wg);
  async_ld_b128(wsb + 16, wg + 8);

  unsigned buf = 0;
  for (int k0 = 0; k0 < Ktot; k0 += 32) {
    wait_async0();        // this wave's DMA for buf done
    __syncthreads();      // every wave's DMA done -> tile complete

    if (k0 + 32 < Ktot) { // overlap: DMA next tile into the other buffer
      const unsigned nb = (buf ^ 1u) * BUFSTRIDE;
      async_ld_b128(asb + nb,      ag + k0 + 32);
      async_ld_b128(asb + nb + 16, ag + k0 + 40);
      async_ld_b128(wsb + nb,      wg + k0 + 32);
      async_ld_b128(wsb + nb + 16, wg + k0 + 40);
      __builtin_prefetch(ag + k0 + 64, 0, 0);   // global_prefetch_b8
      __builtin_prefetch(wg + k0 + 64, 0, 0);
    }

    v16bf a = *(const v16bf*)&As[buf][wv * 16 + ln][lh * 16];
#pragma unroll
    for (int nt = 0; nt < 8; ++nt) {
      v16bf b = *(const v16bf*)&Ws[buf][nt * 16 + ln][lh * 16];
      acc[nt] = wmma_bf16(a, b, acc[nt]);
    }
    __syncthreads();      // reads of buf done before it is refilled next+1
    buf ^= 1u;
  }

  // epilogue: C layout -> VGPR r, lanes 0-15: M=r, lanes 16-31: M=r+8; N=ln
#pragma unroll
  for (int nt = 0; nt < 8; ++nt) {
    const int col = n0 + nt * 16 + ln;
    const float bb = bias[col];
#pragma unroll
    for (int r = 0; r < 8; ++r) {
      const int row = m0 + wv * 16 + r + 8 * lh;
      const float v = acc[nt][r] + bb;
      if (OUT_F32) ((float*)Cout)[(size_t)row * Ntot + col] = v;
      else         ((bf16*)Cout)[(size_t)row * Ntot + col] = (bf16)v;
    }
  }
}

// ---------------------------------------------------------------------------
// V transpose per head: Vt[bh][d][s] = V[b*S+s][h*64+d]  (bf16)
// Gives PV's B-fragment contiguous reads along the key dimension.
// ---------------------------------------------------------------------------
__global__ __launch_bounds__(256)
void transpose_v(const bf16* __restrict__ V, bf16* __restrict__ Vt) {
  size_t i = (size_t)blockIdx.x * 256 + threadIdx.x;   // over B*NH*HD*SEQ
  const int s  = (int)(i & (SEQ - 1));
  const size_t j = i >> 11;                 // / SEQ
  const int d  = (int)(j & (HEAD_DIM - 1));
  const int bh = (int)(j >> 6);             // / HEAD_DIM
  const int b  = bh >> 4;
  const int h  = bh & 15;
  Vt[i] = V[((size_t)(b * SEQ + s)) * H_DIM + h * HEAD_DIM + d];
}

// ---------------------------------------------------------------------------
// Flash attention: one wave (32 threads) per (batch*head, 16-query tile).
// Online softmax over key tiles of 32. Score C-frag row m lives in VGPR
// (m&7) across one 16-lane half -> shfl_xor(width 16) row reductions.
// P (C layout) is relaid to an A fragment through a 1KB LDS tile.
// ---------------------------------------------------------------------------
__global__ __launch_bounds__(32)
void flash_attn(const bf16* __restrict__ Q, const bf16* __restrict__ Kmat,
                const bf16* __restrict__ Vt, const float* __restrict__ mask,
                bf16* __restrict__ Att) {
  __shared__ bf16 Pl[16][32];

  const int lane = threadIdx.x;
  const int ln = lane & 15;
  const int lh = lane >> 4;
  const int bh = blockIdx.x;
  const int b  = bh >> 4;
  const int h  = bh & 15;
  const int q0 = blockIdx.y * 16;

  const bf16* Qh = Q    + (size_t)b * SEQ * H_DIM + (size_t)h * HEAD_DIM;
  const bf16* Kh = Kmat + (size_t)b * SEQ * H_DIM + (size_t)h * HEAD_DIM;
  const bf16* Vh = Vt   + (size_t)bh * HEAD_DIM * SEQ;
  const float* mk = mask + (size_t)b * SEQ;

  // Q fragments (16 queries x 64 head-dim = two K-steps of 32)
  const bf16* qp = Qh + (size_t)(q0 + ln) * H_DIM;
  const v16bf aQ0 = *(const v16bf*)(qp + lh * 16);
  const v16bf aQ1 = *(const v16bf*)(qp + 32 + lh * 16);

  v8f zero = {};
  v8f O[4];
#pragma unroll
  for (int dt = 0; dt < 4; ++dt) O[dt] = zero;
  float mrow[8], lrow[8];
#pragma unroll
  for (int r = 0; r < 8; ++r) { mrow[r] = -1e30f; lrow[r] = 0.0f; }

  for (int k0 = 0; k0 < SEQ; k0 += 32) {
    // --- scores for 32 keys: two 16x16 WMMA outputs ---
    const bf16* kp0 = Kh + (size_t)(k0 + ln) * H_DIM;
    const bf16* kp1 = Kh + (size_t)(k0 + 16 + ln) * H_DIM;
    v16bf bK0a = *(const v16bf*)(kp0 + lh * 16);
    v16bf bK0b = *(const v16bf*)(kp0 + 32 + lh * 16);
    v16bf bK1a = *(const v16bf*)(kp1 + lh * 16);
    v16bf bK1b = *(const v16bf*)(kp1 + 32 + lh * 16);
    v8f s0 = zero, s1 = zero;
    s0 = wmma_bf16(aQ0, bK0a, s0);
    s0 = wmma_bf16(aQ1, bK0b, s0);
    s1 = wmma_bf16(aQ0, bK1a, s1);
    s1 = wmma_bf16(aQ1, bK1b, s1);

    const float madd0 = (1.0f - mk[k0 + ln]) * -10000.0f;
    const float madd1 = (1.0f - mk[k0 + 16 + ln]) * -10000.0f;

    float c[8];
#pragma unroll
    for (int r = 0; r < 8; ++r) {
      float x0 = s0[r] * 0.125f + madd0;   // 1/sqrt(64)
      float x1 = s1[r] * 0.125f + madd1;
      float mx = fmaxf(x0, x1);
#pragma unroll
      for (int off = 8; off >= 1; off >>= 1)
        mx = fmaxf(mx, __shfl_xor(mx, off, 16));
      const float mnew = fmaxf(mrow[r], mx);
      const float p0 = __expf(x0 - mnew);
      const float p1 = __expf(x1 - mnew);
      float ps = p0 + p1;
#pragma unroll
      for (int off = 8; off >= 1; off >>= 1)
        ps += __shfl_xor(ps, off, 16);
      c[r] = __expf(mrow[r] - mnew);
      lrow[r] = lrow[r] * c[r] + ps;
      mrow[r] = mnew;
      // C layout -> LDS [query row][key-in-tile]
      Pl[r + 8 * lh][ln]      = (bf16)p0;
      Pl[r + 8 * lh][16 + ln] = (bf16)p1;
    }
#pragma unroll
    for (int dt = 0; dt < 4; ++dt)
#pragma unroll
      for (int r = 0; r < 8; ++r) O[dt][r] *= c[r];
    __syncthreads();

    // P as A-fragment (16q x 32keys), V^T tiles as B-fragments
    v16bf pA = *(const v16bf*)&Pl[ln][lh * 16];
#pragma unroll
    for (int dt = 0; dt < 4; ++dt) {
      v16bf bV = *(const v16bf*)(Vh + (size_t)(dt * 16 + ln) * SEQ + k0 + lh * 16);
      O[dt] = wmma_bf16(pA, bV, O[dt]);
    }
    __syncthreads();
  }

  // normalize and emit att (concat heads layout [token][H])
  bf16* out = Att + (size_t)b * SEQ * H_DIM + (size_t)h * HEAD_DIM;
#pragma unroll
  for (int r = 0; r < 8; ++r) {
    const float inv = 1.0f / lrow[r];
    const size_t row = (size_t)(q0 + r + 8 * lh);
#pragma unroll
    for (int dt = 0; dt < 4; ++dt)
      out[row * H_DIM + dt * 16 + ln] = (bf16)(O[dt][r] * inv);
  }
}

// ---------------------------------------------------------------------------
// residual add + LayerNorm, one block (256 thr) per token, fp32
// ---------------------------------------------------------------------------
__global__ __launch_bounds__(256)
void residual_layernorm(const float* __restrict__ x, const float* __restrict__ proj,
                        const float* __restrict__ gamma, const float* __restrict__ beta,
                        float* __restrict__ out) {
  const int t = blockIdx.x;
  const int tid = threadIdx.x;
  const float* xr = x + (size_t)t * H_DIM;
  const float* pr = proj + (size_t)t * H_DIM;

  float v[4];
  float sum = 0.0f, sq = 0.0f;
#pragma unroll
  for (int i = 0; i < 4; ++i) {
    const int col = tid + i * 256;
    v[i] = xr[col] + pr[col];
    sum += v[i];
    sq  += v[i] * v[i];
  }
#pragma unroll
  for (int off = 16; off >= 1; off >>= 1) {
    sum += __shfl_xor(sum, off, 32);
    sq  += __shfl_xor(sq, off, 32);
  }
  __shared__ float s1[8], s2[8];
  if ((tid & 31) == 0) { s1[tid >> 5] = sum; s2[tid >> 5] = sq; }
  __syncthreads();
  float ts = 0.0f, tq = 0.0f;
#pragma unroll
  for (int i = 0; i < 8; ++i) { ts += s1[i]; tq += s2[i]; }
  const float mean = ts * (1.0f / H_DIM);
  const float var  = tq * (1.0f / H_DIM) - mean * mean;
  const float rstd = rsqrtf(var + 1e-12f);
#pragma unroll
  for (int i = 0; i < 4; ++i) {
    const int col = tid + i * 256;
    out[(size_t)t * H_DIM + col] = gamma[col] * ((v[i] - mean) * rstd) + beta[col];
  }
}

// ---------------------------------------------------------------------------
extern "C" void kernel_launch(void* const* d_in, const int* in_sizes, int n_in,
                              void* d_out, int out_size, void* d_ws, size_t ws_size,
                              hipStream_t stream) {
  (void)in_sizes; (void)n_in; (void)out_size; (void)ws_size;

  const float* pre_out = (const float*)d_in[0];
  const float* mask    = (const float*)d_in[1];
  const float* Wq = (const float*)d_in[2];
  const float* bq = (const float*)d_in[3];
  const float* Wk = (const float*)d_in[4];
  const float* bk = (const float*)d_in[5];
  const float* Wv = (const float*)d_in[6];
  const float* bv = (const float*)d_in[7];
  const float* Wo = (const float*)d_in[8];
  const float* bo = (const float*)d_in[9];
  const float* lnw = (const float*)d_in[10];
  const float* lnb = (const float*)d_in[11];
  float* out = (float*)d_out;

  // workspace carve (~72 MiB total)
  char* w = (char*)d_ws;
  size_t off = 0;
  auto carve = [&](size_t bytes) -> void* {
    void* p = w + off;
    off += (bytes + 255) & ~(size_t)255;
    return p;
  };
  const size_t TOK_H = (size_t)TOKENS * H_DIM;            // 4M elems
  const size_t W_ELEMS = (size_t)H_DIM * H_DIM;           // 1M elems
  bf16* Xb  = (bf16*)carve(TOK_H * 2);
  bf16* Wqb = (bf16*)carve(W_ELEMS * 2);
  bf16* Wkb = (bf16*)carve(W_ELEMS * 2);
  bf16* Wvb = (bf16*)carve(W_ELEMS * 2);
  bf16* Wob = (bf16*)carve(W_ELEMS * 2);
  bf16* Qb  = (bf16*)carve(TOK_H * 2);
  bf16* Kb  = (bf16*)carve(TOK_H * 2);
  bf16* Vb  = (bf16*)carve(TOK_H * 2);
  bf16* Vtb = (bf16*)carve(TOK_H * 2);
  bf16* Ab  = (bf16*)carve(TOK_H * 2);
  float* Ob = (float*)carve(TOK_H * 4);

  // 1) downconvert activations + weights to bf16
  {
    const int n4 = (int)(TOK_H / 4);
    cvt_f32_bf16<<<(n4 + 255) / 256, 256, 0, stream>>>(pre_out, Xb, n4);
    const int w4 = (int)(W_ELEMS / 4);
    const int wg = (w4 + 255) / 256;
    cvt_f32_bf16<<<wg, 256, 0, stream>>>(Wq, Wqb, w4);
    cvt_f32_bf16<<<wg, 256, 0, stream>>>(Wk, Wkb, w4);
    cvt_f32_bf16<<<wg, 256, 0, stream>>>(Wv, Wvb, w4);
    cvt_f32_bf16<<<wg, 256, 0, stream>>>(Wo, Wob, w4);
  }

  // 2) QKV projections (bf16 WMMA GEMMs, async-LDS double-buffered)
  dim3 gg(TOKENS / 128, H_DIM / 128);
  gemm_xwT<false><<<gg, 256, 0, stream>>>(Xb, Wqb, bq, Qb, H_DIM, H_DIM);
  gemm_xwT<false><<<gg, 256, 0, stream>>>(Xb, Wkb, bk, Kb, H_DIM, H_DIM);
  gemm_xwT<false><<<gg, 256, 0, stream>>>(Xb, Wvb, bv, Vb, H_DIM, H_DIM);

  // 3) V^T per head for the PV matmul
  transpose_v<<<(int)(TOK_H / 256), 256, 0, stream>>>(Vb, Vtb);

  // 4) flash attention
  dim3 ga(BATCH * NHEAD, SEQ / 16);
  flash_attn<<<ga, 32, 0, stream>>>(Qb, Kb, Vtb, mask, Ab);

  // 5) output projection (fp32 out)
  gemm_xwT<true><<<gg, 256, 0, stream>>>(Ab, Wob, bo, Ob, H_DIM, H_DIM);

  // 6) residual + LayerNorm
  residual_layernorm<<<TOKENS, 256, 0, stream>>>(pre_out, Ob, lnw, lnb, out);
}
